// MultiHeadAttention_16604343567227
// MI455X (gfx1250) — matrix operand order repaired and checked by reading, not compiled, not run on hardware
//
#include <hip/hip_runtime.h>
#include <stdint.h>

// ---------------------------------------------------------------------------
// Multi-head causal attention for MI455X (gfx1250, wave32, WMMA).
// All matmuls use v_wmma_f32_16x16x32_bf16; softmax/accum in fp32.
// Tile staging uses the Tensor Data Mover (tensor_load_to_lds, 6-arg
// clang-23 form): one wave DMAs each 2D tile into bank-padded LDS (TDM
// hardware padding), tracked via TENSORcnt.
// Fallback path: vectorized b128 register staging.
// ---------------------------------------------------------------------------

#define BB 4
#define TT 2048
#define CC 1024
#define NH 16
#define DH 64

typedef __attribute__((ext_vector_type(16))) __bf16 v16bf;
typedef __attribute__((ext_vector_type(8)))  float  v8f;

struct __align__(16) U4 { unsigned x, y, z, w; };

union BF16x16 { v16bf v; unsigned u[8]; U4 q[2]; };

__device__ __forceinline__ unsigned short f2bf(float f) {
  unsigned u = __float_as_uint(f);
  u = u + 0x7FFFu + ((u >> 16) & 1u);   // round-to-nearest-even
  return (unsigned short)(u >> 16);
}

// wave-local LDS fence: same-wave DS ops complete in order; just wait DScnt.
__device__ __forceinline__ void wave_lds_fence() {
  asm volatile("s_wait_dscnt 0x0" ::: "memory");
}

// ------------------------- Tensor Data Mover helpers -----------------------
#if __has_builtin(__builtin_amdgcn_tensor_load_to_lds)
#define HAVE_TDM 1
typedef unsigned int u32x4 __attribute__((ext_vector_type(4)));
typedef int          i32x8 __attribute__((ext_vector_type(8)));
typedef int          i32x4 __attribute__((ext_vector_type(4)));

// 2D bf16 tile load: tile_rows x row_elems from global (row stride
// stride_elems) into LDS at lds_off, with TDM padding of 4 DWORDs after
// every (1<<pad_ivl_code)*2 DWORDs -- matches our bank-padded LDS tiles.
__device__ __forceinline__ void tdm_load_2d(unsigned lds_off,
                                            const void* gaddr,
                                            unsigned row_elems,
                                            unsigned tile_rows,
                                            unsigned stride_elems,
                                            unsigned pad_ivl_code) {
  unsigned long long ga = (unsigned long long)gaddr;
  u32x4 g0;
  g0[0] = 1u;                                            // count=1, user mode
  g0[1] = lds_off;                                       // lds_addr (bytes)
  g0[2] = (unsigned)(ga & 0xFFFFFFFFu);                  // global_addr lo
  g0[3] = (unsigned)((ga >> 32) & 0x01FFFFFFu) | (2u << 30);  // hi | type=2
  i32x8 g1;
  g1[0] = (int)((1u << 16)                 // data_size = 2 bytes (bf16)
              | (1u << 20)                 // pad_enable
              | (pad_ivl_code << 22)       // pad_interval
              | (3u << 25));               // pad_amount = 4 DWORDs
  g1[1] = (int)((row_elems & 0xFFFFu) << 16);            // tensor_dim0[15:0]
  g1[2] = (int)(((row_elems >> 16) & 0xFFFFu)            // tensor_dim0[31:16]
              | ((tile_rows & 0xFFFFu) << 16));          // tensor_dim1[15:0]
  g1[3] = (int)(((tile_rows >> 16) & 0xFFFFu)            // tensor_dim1[31:16]
              | ((row_elems & 0xFFFFu) << 16));          // tile_dim0
  g1[4] = (int)(tile_rows & 0xFFFFu);                    // tile_dim1 (dim2=0)
  g1[5] = (int)stride_elems;                             // dim0_stride[31:0]
  g1[6] = 0;                                             // stride hi bits
  g1[7] = 0;
  i32x4 z4 = {0, 0, 0, 0};
  i32x8 z8 = {0, 0, 0, 0, 0, 0, 0, 0};
  __builtin_amdgcn_tensor_load_to_lds(g0, g1, z4, z4, z8, 0);
}

__device__ __forceinline__ void wait_tensor0() {
#if __has_builtin(__builtin_amdgcn_s_wait_tensorcnt)
  __builtin_amdgcn_s_wait_tensorcnt(0);
#else
  asm volatile("s_wait_tensorcnt 0x0" ::: "memory");
#endif
}
#else
#define HAVE_TDM 0
#endif

__device__ __forceinline__ unsigned lds_offset_of(const void* p) {
  return (unsigned)(size_t)p;   // low 32 bits of generic addr = LDS offset
}

// ------------------------- fp32 -> bf16 convert ---------------------------
__global__ void f32_to_bf16_kernel(const float* __restrict__ in,
                                   unsigned short* __restrict__ out, int n2) {
  int i = blockIdx.x * blockDim.x + threadIdx.x;
  if (i >= n2) return;
  float2 f = ((const float2*)in)[i];
  unsigned p = (unsigned)f2bf(f.x) | ((unsigned)f2bf(f.y) << 16);
  ((unsigned*)out)[i] = p;
}

// ----------------- weight transpose + convert: W[k][n] -> Wt[n][k] --------
__global__ void transpose_w_kernel(const float* __restrict__ W,
                                   unsigned short* __restrict__ Wt) {
  __shared__ float tile[32][33];
  const int n0 = blockIdx.x * 32, k0 = blockIdx.y * 32;
  const int tx = threadIdx.x, ty = threadIdx.y;   // 32 x 8
#pragma unroll
  for (int i = 0; i < 32; i += 8)
    tile[ty + i][tx] = W[(size_t)(k0 + ty + i) * CC + n0 + tx];
  __syncthreads();
#pragma unroll
  for (int i = 0; i < 32; i += 8)
    Wt[(size_t)(n0 + ty + i) * CC + k0 + tx] = f2bf(tile[tx][ty + i]);
}

// ------------------------------- GEMM --------------------------------------
// Y[m][n] = sum_k A[m][k] * Bt[n][k] + bias[n]
// Block: 128 threads = 4 waves as 2(M)x2(N); block tile 64(M) x 128(N);
// wave tile 32x64 -> 8 WMMAs per K=32 step from 12 b128 LDS loads.
// LDS rows: 16 data uints + 4 pad (stride 20: 16B aligned, bank-spread).
// Double-buffered; TDM stages the next K-slab while WMMAs run.
template <bool OUT_F32>
__global__ void __launch_bounds__(128)
gemm_wmma_kernel(const unsigned short* __restrict__ A,
                 const unsigned short* __restrict__ Bt,
                 const float* __restrict__ bias,
                 float* __restrict__ outF,
                 unsigned short* __restrict__ outB,
                 int st_t, int st_d) {
  __shared__ unsigned lds_a[2][64 * 20];    // 64 rows x 32 bf16, stride 20
  __shared__ unsigned lds_b[2][128 * 20];   // 128 rows x 32 bf16

  const int m0 = blockIdx.x * 64, n0 = blockIdx.y * 128;
  const int tid = threadIdx.x;
  const int w = tid >> 5, lane = tid & 31;
  const int g = lane >> 4, lh = lane & 15;
  const int wm = w >> 1, wn = w & 1;

  const unsigned* Au = (const unsigned*)A;
  const unsigned* Bu = (const unsigned*)Bt;

#if HAVE_TDM
  auto issue_tiles = [&](int kt, int buf) {
    if (w == 0) {
      const int kk = kt * 16;
      tdm_load_2d(lds_offset_of(&lds_a[buf][0]),
                  Au + (size_t)m0 * (CC / 2) + kk, 32, 64, CC, 3);
      tdm_load_2d(lds_offset_of(&lds_b[buf][0]),
                  Bu + (size_t)n0 * (CC / 2) + kk, 32, 128, CC, 3);
    }
  };
#else
  const int srow = tid >> 2, squad = tid & 3;   // staging coords
  U4 sa[2], sb[4];
  auto load_tiles = [&](int kt) {
    const int kk = kt * 16;
#pragma unroll
    for (int j = 0; j < 2; ++j)
      sa[j] = *(const U4*)(Au + (size_t)(m0 + srow + j * 32) * (CC / 2) + kk + squad * 4);
#pragma unroll
    for (int j = 0; j < 4; ++j)
      sb[j] = *(const U4*)(Bu + (size_t)(n0 + srow + j * 32) * (CC / 2) + kk + squad * 4);
  };
  auto store_tiles = [&](int buf) {
#pragma unroll
    for (int j = 0; j < 2; ++j)
      *(U4*)&lds_a[buf][(srow + j * 32) * 20 + squad * 4] = sa[j];
#pragma unroll
    for (int j = 0; j < 4; ++j)
      *(U4*)&lds_b[buf][(srow + j * 32) * 20 + squad * 4] = sb[j];
  };
#endif

  v8f c[2][4] = {};

#if HAVE_TDM
  issue_tiles(0, 0);
  wait_tensor0();
#else
  load_tiles(0);
  store_tiles(0);
#endif
  __syncthreads();

  const int KT = CC / 32;
  for (int kt = 0; kt < KT; ++kt) {
    const int cur = kt & 1;
#if HAVE_TDM
    if (kt + 1 < KT) issue_tiles(kt + 1, cur ^ 1);
#else
    if (kt + 1 < KT) load_tiles(kt + 1);
#endif

    // A fragments: row = lane%16; K-pairs [g*4..g*4+3] and [g*4+8..g*4+11]
    BF16x16 a[2];
#pragma unroll
    for (int i = 0; i < 2; ++i) {
      const unsigned* pa = &lds_a[cur][(wm * 32 + i * 16 + lh) * 20 + g * 4];
      a[i].q[0] = *(const U4*)pa;
      a[i].q[1] = *(const U4*)(pa + 8);
    }
#pragma unroll
    for (int f = 0; f < 4; ++f) {
      // B fragment: col = lane%16; K-pairs [g*4..g*4+3] and [g*4+8..g*4+11]
      BF16x16 bfr;
      const unsigned* pb = &lds_b[cur][(wn * 64 + f * 16 + lh) * 20 + g * 4];
      bfr.q[0] = *(const U4*)pb;
      bfr.q[1] = *(const U4*)(pb + 8);
#pragma unroll
      for (int i = 0; i < 2; ++i)
        c[i][f] = __builtin_amdgcn_wmma_f32_16x16x32_bf16(
            false, a[i].v, false, bfr.v, (short)0, c[i][f], false, false);
    }

#if HAVE_TDM
    wait_tensor0();
#else
    if (kt + 1 < KT) store_tiles(cur ^ 1);
#endif
    __syncthreads();
  }

  // epilogue: bias + store
#pragma unroll
  for (int i = 0; i < 2; ++i)
#pragma unroll
    for (int f = 0; f < 4; ++f) {
      const int nn = n0 + wn * 64 + f * 16 + lh;
      const float bvv = bias[nn];
#pragma unroll
      for (int r = 0; r < 8; ++r) {
        const int mm = m0 + wm * 32 + i * 16 + r + g * 8;
        const float val = c[i][f][r] + bvv;
        if (OUT_F32) {
          outF[(size_t)mm * CC + nn] = val;
        } else {
          const int b = mm / TT, t = mm % TT;
          const int h = nn >> 6, d = nn & 63;
          size_t off = (size_t)(b * NH + h) * (TT * DH) +
                       (size_t)t * st_t + (size_t)d * st_d;
          outB[off] = f2bf(val);
        }
      }
    }
}

// --------------------------- flash attention -------------------------------
// grid: (T/64, B*NH), block 128 (4 waves). Block owns 64 q rows of one head.
// Qh,Kh: [bh][T][64] bf16 ; Vt: [bh][64][T] bf16 ; AO: [B*T][C] bf16.
// Double-buffered K/V tiles via TDM; P round-trip is wave-private.
__global__ void __launch_bounds__(128)
attn_kernel(const unsigned short* __restrict__ Qh,
            const unsigned short* __restrict__ Kh,
            const unsigned short* __restrict__ Vt,
            unsigned short* __restrict__ AO) {
  __shared__ unsigned k_tile[2][64 * 36];    // [j][d-pairs], 32 data + 4 pad
  __shared__ unsigned v_tile[2][64 * 36];    // [d][j-pairs]
  __shared__ unsigned p_buf[4 * 16 * 36];    // per-wave P (16 x 64 bf16)

  const int q0 = blockIdx.x * 64;
  const int bh = blockIdx.y;
  const int b = bh >> 4, h = bh & 15;
  const int tid = threadIdx.x, w = tid >> 5, lane = tid & 31;
  const int g = lane >> 4, lh = lane & 15;

  const unsigned* Qu = (const unsigned*)Qh + (size_t)bh * TT * 32;
  const unsigned* Ku = (const unsigned*)Kh + (size_t)bh * TT * 32;
  const unsigned* Vu = (const unsigned*)Vt + (size_t)bh * DH * (TT / 2);

  // preload Q fragments (rows q0 + w*16 + lane%16, two K=32 steps over d=64)
  BF16x16 qa[2];
  {
    const int qrow = q0 + w * 16 + lh;
#pragma unroll
    for (int ks = 0; ks < 2; ++ks) {
      const unsigned* pq = Qu + (size_t)qrow * 32 + ks * 16 + g * 4;
      qa[ks].q[0] = *(const U4*)pq;
      qa[ks].q[1] = *(const U4*)(pq + 8);
    }
  }

  v8f o[4] = {};
  float m_r[8], l_r[8];
#pragma unroll
  for (int r = 0; r < 8; ++r) { m_r[r] = -1e30f; l_r[r] = 0.f; }

  unsigned* pw = p_buf + w * 16 * 36;
  unsigned short* pws = (unsigned short*)pw;

#if HAVE_TDM
  auto issue_kv = [&](int j0n, int buf) {
    if (w == 0) {
      tdm_load_2d(lds_offset_of(&k_tile[buf][0]),
                  Ku + (size_t)j0n * 32, 64, 64, 64, 4);
      tdm_load_2d(lds_offset_of(&v_tile[buf][0]),
                  Vu + j0n / 2, 64, 64, TT, 4);
    }
  };
#else
  const int srow = tid >> 3, squad = tid & 7;   // staging coords
  U4 sk[4], sv[4];
  auto load_kv = [&](int j0n) {
#pragma unroll
    for (int j = 0; j < 4; ++j) {
      const int row = srow + j * 16;
      sk[j] = *(const U4*)(Ku + (size_t)(j0n + row) * 32 + squad * 4);
      sv[j] = *(const U4*)(Vu + (size_t)row * (TT / 2) + j0n / 2 + squad * 4);
    }
  };
  auto store_kv = [&](int buf) {
#pragma unroll
    for (int j = 0; j < 4; ++j) {
      const int row = srow + j * 16;
      *(U4*)&k_tile[buf][row * 36 + squad * 4] = sk[j];
      *(U4*)&v_tile[buf][row * 36 + squad * 4] = sv[j];
    }
  };
#endif

  const int ntiles = blockIdx.x + 1;   // causal: only tiles up to diagonal
#if HAVE_TDM
  issue_kv(0, 0);
  wait_tensor0();
#else
  load_kv(0);
  store_kv(0);
#endif
  __syncthreads();

  for (int jt = 0; jt < ntiles; ++jt) {
    const int cur = jt & 1;
    const int j0 = jt * 64;
#if HAVE_TDM
    if (jt + 1 < ntiles) issue_kv(j0 + 64, cur ^ 1);
#else
    if (jt + 1 < ntiles) load_kv(j0 + 64);
#endif

    // scores S = Q K^T  (B[d][j] = Kh[j][d]; K-pairs contiguous in k_tile row)
    v8f s[4] = {};
#pragma unroll
    for (int f = 0; f < 4; ++f) {
      const int jr = f * 16 + lh;
#pragma unroll
      for (int ks = 0; ks < 2; ++ks) {
        BF16x16 kb;
        const unsigned* pb = &k_tile[cur][jr * 36 + ks * 16 + g * 4];
        kb.q[0] = *(const U4*)pb;
        kb.q[1] = *(const U4*)(pb + 8);
        s[f] = __builtin_amdgcn_wmma_f32_16x16x32_bf16(
            false, qa[ks].v, false, kb.v, (short)0, s[f], false, false);
      }
    }

    // scale by 1/sqrt(64) and apply causal mask
#pragma unroll
    for (int f = 0; f < 4; ++f) {
      const int col = j0 + f * 16 + lh;
#pragma unroll
      for (int r = 0; r < 8; ++r) {
        const int row = q0 + w * 16 + r + g * 8;
        float svv = s[f][r] * 0.125f;
        if (col > row) svv = -1e9f;
        s[f][r] = svv;
      }
    }

    // online softmax (row reductions across 16-lane halves, wave32)
    float scale_r[8];
#pragma unroll
    for (int r = 0; r < 8; ++r) {
      float t = fmaxf(fmaxf(s[0][r], s[1][r]), fmaxf(s[2][r], s[3][r]));
      t = fmaxf(t, __shfl_xor(t, 1, 32));
      t = fmaxf(t, __shfl_xor(t, 2, 32));
      t = fmaxf(t, __shfl_xor(t, 4, 32));
      t = fmaxf(t, __shfl_xor(t, 8, 32));
      const float mn = fmaxf(m_r[r], t);
      const float sc = __expf(m_r[r] - mn);
      m_r[r] = mn; scale_r[r] = sc;
      float rs = 0.f;
#pragma unroll
      for (int f = 0; f < 4; ++f) {
        float p = __expf(s[f][r] - mn);
        s[f][r] = p; rs += p;
      }
      rs += __shfl_xor(rs, 1, 32);
      rs += __shfl_xor(rs, 2, 32);
      rs += __shfl_xor(rs, 4, 32);
      rs += __shfl_xor(rs, 8, 32);
      l_r[r] = l_r[r] * sc + rs;
    }

    // C-layout -> A-layout via wave-private LDS; rescale O accumulators
#pragma unroll
    for (int f = 0; f < 4; ++f) {
#pragma unroll
      for (int r = 0; r < 8; ++r) {
        pws[(r + g * 8) * 72 + f * 16 + lh] = f2bf(s[f][r]);
        o[f][r] *= scale_r[r];
      }
    }
    wave_lds_fence();   // same-wave DS ops are in-order; DScnt==0 suffices

    // O += P V  (B[j][d] = Vt[d][j]; K-pairs contiguous in v_tile row)
#pragma unroll
    for (int ks = 0; ks < 2; ++ks) {
      BF16x16 pa;
      const unsigned* pp = pw + lh * 36 + ks * 16 + g * 4;
      pa.q[0] = *(const U4*)pp;
      pa.q[1] = *(const U4*)(pp + 8);
#pragma unroll
      for (int f = 0; f < 4; ++f) {
        BF16x16 vb;
        const unsigned* pb = &v_tile[cur][(f * 16 + lh) * 36 + ks * 16 + g * 4];
        vb.q[0] = *(const U4*)pb;
        vb.q[1] = *(const U4*)(pb + 8);
        o[f] = __builtin_amdgcn_wmma_f32_16x16x32_bf16(
            false, pa.v, false, vb.v, (short)0, o[f], false, false);
      }
    }

#if HAVE_TDM
    wait_tensor0();
#else
    if (jt + 1 < ntiles) store_kv(cur ^ 1);
#endif
    __syncthreads();
  }

  // normalize and write [B*T][C] bf16 for the output projection
#pragma unroll
  for (int f = 0; f < 4; ++f) {
#pragma unroll
    for (int r = 0; r < 8; ++r) {
      const int t = q0 + w * 16 + r + g * 8;
      const int d = f * 16 + lh;
      const float val = o[f][r] / l_r[r];
      AO[((size_t)(b * TT + t)) * CC + h * DH + d] = f2bf(val);
    }
  }
}

// ------------------------------ launcher -----------------------------------
extern "C" void kernel_launch(void* const* d_in, const int* in_sizes, int n_in,
                              void* d_out, int out_size, void* d_ws, size_t ws_size,
                              hipStream_t stream) {
  const float* q  = (const float*)d_in[0];
  const float* k  = (const float*)d_in[1];
  const float* v  = (const float*)d_in[2];
  const float* wq = (const float*)d_in[3];
  const float* bq = (const float*)d_in[4];
  const float* wk = (const float*)d_in[5];
  const float* bk = (const float*)d_in[6];
  const float* wv = (const float*)d_in[7];
  const float* bv = (const float*)d_in[8];
  const float* wo = (const float*)d_in[9];
  const float* bo = (const float*)d_in[10];
  float* out = (float*)d_out;

  unsigned short* ws = (unsigned short*)d_ws;
  const size_t WSZ = (size_t)CC * CC;        // 1M elems per weight
  const size_t XSZ = (size_t)BB * TT * CC;   // 8M elems per activation
  unsigned short* WTQ = ws;
  unsigned short* WTK = WTQ + WSZ;
  unsigned short* WTV = WTK + WSZ;
  unsigned short* WTO = WTV + WSZ;
  unsigned short* XQ  = WTO + WSZ;
  unsigned short* XK  = XQ + XSZ;
  unsigned short* XV  = XK + XSZ;
  unsigned short* QH  = XV + XSZ;
  unsigned short* KH  = QH + XSZ;
  unsigned short* VT  = KH + XSZ;
  unsigned short* AO  = VT + XSZ;

  // 1) convert activations to bf16
  const int n2 = (int)(XSZ / 2);
  f32_to_bf16_kernel<<<dim3((n2 + 255) / 256), dim3(256), 0, stream>>>(q, XQ, n2);
  f32_to_bf16_kernel<<<dim3((n2 + 255) / 256), dim3(256), 0, stream>>>(k, XK, n2);
  f32_to_bf16_kernel<<<dim3((n2 + 255) / 256), dim3(256), 0, stream>>>(v, XV, n2);

  // 2) transpose+convert weights: W[k][n] -> Wt[n][k] bf16
  dim3 tg(CC / 32, CC / 32), tb(32, 8);
  transpose_w_kernel<<<tg, tb, 0, stream>>>(wq, WTQ);
  transpose_w_kernel<<<tg, tb, 0, stream>>>(wk, WTK);
  transpose_w_kernel<<<tg, tb, 0, stream>>>(wv, WTV);
  transpose_w_kernel<<<tg, tb, 0, stream>>>(wo, WTO);

  // 3) QKV projections (Q/K head-major [b,h,t,d]; V transposed [b,h,d,t])
  dim3 gg(BB * TT / 64, CC / 128), gb(128);
  gemm_wmma_kernel<false><<<gg, gb, 0, stream>>>(XQ, WTQ, bq, nullptr, QH, DH, 1);
  gemm_wmma_kernel<false><<<gg, gb, 0, stream>>>(XK, WTK, bk, nullptr, KH, DH, 1);
  gemm_wmma_kernel<false><<<gg, gb, 0, stream>>>(XV, WTV, bv, nullptr, VT, 1, TT);

  // 4) causal flash attention
  attn_kernel<<<dim3(TT / 64, BB * NH), dim3(128), 0, stream>>>(QH, KH, VT, AO);

  // 5) output projection (fp32 out)
  gemm_wmma_kernel<true><<<gg, gb, 0, stream>>>(AO, WTO, bo, out, nullptr, 0, 0);
}
